// LF_WXformerCHUAN_29326036697186
// MI455X (gfx1250) — compile-verified, matrix operand order checked by reading, not run on hardware
//
#include <hip/hip_runtime.h>
#include <hip/hip_bf16.h>

// MI455X (gfx1250): WMMA-dense pipeline. f16 operands pre-packed so every
// fragment is two 16B loads; flash attention uses LDS-staged K/V chunks
// (GLOBAL_LOAD_ASYNC_TO_LDS_B128 when available) and a max-free exp2-domain
// softmax (valid here: logits analytically bounded, decay mask <= 0).

typedef __attribute__((ext_vector_type(16))) _Float16 v16h;
typedef __attribute__((ext_vector_type(8)))  _Float16 v8h;
typedef __attribute__((ext_vector_type(8)))  float    v8f;
typedef int v4i __attribute__((vector_size(16)));
typedef __attribute__((address_space(1))) v4i* gv4i_p;   // global (device) AS
typedef __attribute__((address_space(3))) v4i* sv4i_p;   // LDS AS

#define EMBED    256
#define HEADS    8
#define HEAD_DIM 32
#define BATCH    2
#define IMG      48
#define NTOK     (IMG * IMG)      // 2304
#define ROWS     (BATCH * NTOK)   // 4608
#define MATSZ    (ROWS * EMBED)   // 1179648
#define NCHUNK   (NTOK / 32)      // 72

#if defined(__has_builtin)
#if __has_builtin(__builtin_amdgcn_global_load_async_to_lds_b128) && \
    __has_builtin(__builtin_amdgcn_s_wait_asynccnt)
#define USE_ASYNC 1
#endif
#endif

// Build a 16-bit WMMA A/B fragment (ISA 7.12.2) from a contiguous row of
// >=32 f16 elements: element i <- p[(i>>3)*16 + hi*8 + (i&7)].
// Exactly two 16-byte vector loads per lane. p must be 16B aligned.
__device__ __forceinline__ v16h frag_row(const _Float16* __restrict__ p, int lane) {
  const int hi8 = ((lane >> 4) & 1) << 3;
  v8h a0 = *(const v8h*)(p + hi8);
  v8h a1 = *(const v8h*)(p + 16 + hi8);
  v16h r;
  #pragma unroll
  for (int j = 0; j < 8; ++j) { r[j] = a0[j]; r[8 + j] = a1[j]; }
  return r;
}

// ---------------------------------------------------------------------------
// Kernel 0: pack X -> f16, and Wq/Wk/Wv/Wo -> f16 transposed (Wt[col][k]).
// ---------------------------------------------------------------------------
__global__ __launch_bounds__(256) void pack_kernel(
    const float* __restrict__ X,
    const float* __restrict__ Wq, const float* __restrict__ Wk,
    const float* __restrict__ Wv, const float* __restrict__ Wo,
    _Float16* __restrict__ Xh, _Float16* __restrict__ Wt4)
{
  const int t = blockIdx.x * 256 + threadIdx.x;
  if (t < MATSZ) {
    Xh[t] = (_Float16)X[t];
  } else {
    const int u   = t - MATSZ;          // < 4*65536
    const int w   = u >> 16;
    const int r   = u & 65535;
    const int col = r & 255;            // consecutive -> coalesced read
    const int k   = r >> 8;
    const float* W = (w == 0) ? Wq : (w == 1) ? Wk : (w == 2) ? Wv : Wo;
    Wt4[(w << 16) + col * 256 + k] = (_Float16)W[k * 256 + col];
  }
}

// ---------------------------------------------------------------------------
// Kernel 1: fused QKV GEMM + bias + rotary + k-scaling (log2e folded into k).
// One wave = 16x32 output tile. Writes Qh/Kh f16 [bh][n][d]; Vf f32; Vt f16^T.
// ---------------------------------------------------------------------------
__global__ __launch_bounds__(256) void qkv_rotary_kernel(
    const _Float16* __restrict__ Xh, const _Float16* __restrict__ Wt4,
    const float* __restrict__ bq, const float* __restrict__ bk,
    const float* __restrict__ bv,
    _Float16* __restrict__ Qh, _Float16* __restrict__ Kh,
    float* __restrict__ Vf, _Float16* __restrict__ Vt)
{
  const int lane  = threadIdx.x & 31;
  const int wid   = threadIdx.x >> 5;
  const int wtile = blockIdx.x * 8 + wid;     // 0..6911
  const int tn    = wtile % 24;               // 768/32 col-tiles
  const int tm    = wtile / 24;               // 4608/16 row-tiles
  const int m0    = tm * 16;
  const int n0    = tn * 32;
  const int ln    = lane & 15;
  const int hi    = (lane >> 4) & 1;
  const int which = n0 >> 8;                  // 0=Q 1=K 2=V (uniform per wave)
  const int nbase = n0 & 255;

  const _Float16* __restrict__ Wt = Wt4 + (which << 16);
  const float* __restrict__ bias  = (which == 0) ? bq : (which == 1) ? bk : bv;

  const _Float16* __restrict__ arow  = Xh + (size_t)(m0 + ln) * 256;
  const _Float16* __restrict__ bcol0 = Wt + (size_t)(nbase + ln) * 256;
  const _Float16* __restrict__ bcol1 = Wt + (size_t)(nbase + 16 + ln) * 256;

  v8f acc0 = {}, acc1 = {};
  #pragma unroll
  for (int k0 = 0; k0 < 256; k0 += 32) {
    const v16h a  = frag_row(arow + k0, lane);
    const v16h b0 = frag_row(bcol0 + k0, lane);
    const v16h b1 = frag_row(bcol1 + k0, lane);
    acc0 = __builtin_amdgcn_wmma_f32_16x16x32_f16(false, a, false, b0,
                                                  (short)0, acc0, false, false);
    acc1 = __builtin_amdgcn_wmma_f32_16x16x32_f16(false, a, false, b1,
                                                  (short)0, acc1, false, false);
  }

  // HEAD_DIM^-0.5 * log2(e): exp2-domain softmax downstream
  const float scaling = 0.17677669529663687f * 1.4426950408889634f;

  #pragma unroll
  for (int t = 0; t < 2; ++t) {
    const int   col  = nbase + t * 16 + ln;   // 0..255 within this matrix
    const int   head = col >> 5;
    const int   d    = col & 31;
    const float bval = bias[col];
    // angle[d] = 10000^(-(d>>1)/15) = exp2(-(d>>1)*log2(10000)/15)
    const float freq = exp2f((float)(d >> 1) * -0.8858474916583548f);
    #pragma unroll
    for (int r = 0; r < 8; ++r) {
      const int   row  = m0 + r + 8 * hi;     // C/D layout: M = r + 8*hi
      const int   bimg = row / NTOK;
      const int   pos  = row - bimg * NTOK;
      const int   bh   = bimg * HEADS + head;
      const float val  = (t ? acc1[r] : acc0[r]) + bval;
      const size_t dst = ((size_t)bh * NTOK + pos) * HEAD_DIM + d;
      if (which == 2) {
        Vf[dst] = val;
        Vt[((size_t)bh * HEAD_DIM + d) * NTOK + pos] = (_Float16)val;
      } else {
        const float partner = __shfl_xor(val, 1, 32);   // col^1 lives in lane^1
        float s, c;
        __sincosf((float)pos * freq, &s, &c);
        const float rot = (d & 1) ? partner : -partner;
        const float out = val * c + rot * s;
        if (which == 0) Qh[dst] = (_Float16)out;
        else            Kh[dst] = (_Float16)(out * scaling);
      }
    }
  }
}

// ---------------------------------------------------------------------------
// Kernel 2: depthwise 5x5 conv (lepe), SAME padding.
// In: Vf f32 [bh][n][d].  Out: Lp f32 [row][c]  (row = b*NTOK+n, c = head*32+d)
// ---------------------------------------------------------------------------
__global__ __launch_bounds__(256) void lepe_kernel(
    const float* __restrict__ Vf, const float* __restrict__ Wc,
    const float* __restrict__ bc, float* __restrict__ Lp)
{
  const int t  = blockIdx.x * 256 + threadIdx.x;   // < MATSZ
  const int d  = t & 31;
  const int n  = (t >> 5) % NTOK;
  const int bh = t / (NTOK * 32);
  const int c  = (bh & 7) * 32 + d;
  const int h  = n / IMG, w = n % IMG;
  float acc = bc[c];
  #pragma unroll
  for (int ky = 0; ky < 5; ++ky) {
    const int hh = h + ky - 2;
    if (hh < 0 || hh >= IMG) continue;
    #pragma unroll
    for (int kx = 0; kx < 5; ++kx) {
      const int ww = w + kx - 2;
      if (ww < 0 || ww >= IMG) continue;
      acc += Vf[(bh * NTOK + hh * IMG + ww) * 32 + d] * Wc[c * 25 + ky * 5 + kx];
    }
  }
  Lp[(size_t)((bh >> 3) * NTOK + n) * 256 + c] = acc;
}

// ---------------------------------------------------------------------------
// Kernel 3: flash attention, max-free exp2 softmax, LDS-staged K/V chunks
// (async copy to LDS when available), fused lepe add.
// ---------------------------------------------------------------------------
__global__ __launch_bounds__(256) void flash_attn_kernel(
    const _Float16* __restrict__ Qh, const _Float16* __restrict__ Kh,
    const _Float16* __restrict__ Vt, const float* __restrict__ Lp,
    _Float16* __restrict__ Ah)
{
  __shared__ __align__(16) _Float16 Kc[2][32][32];     // key rows x dims
  __shared__ __align__(16) _Float16 Vc[2][32][32];     // dims x key cols (V^T)
  __shared__ __align__(16) _Float16 pbuf[8][16][32];   // per-wave P staging

  const int tid  = threadIdx.x;
  const int lane = tid & 31;
  const int wid  = tid >> 5;
  const int bh   = blockIdx.x / 18;          // 0..15
  const int qblk = blockIdx.x % 18;
  const int q0   = qblk * 128 + wid * 16;
  const int head = bh & 7;
  const int bimg = bh >> 3;
  const int ln   = lane & 15;
  const int hi   = (lane >> 4) & 1;

  // decay (<=0) in log2 domain
  const float decay2 =
      logf(1.0f - exp2f(-1.0f - 3.0f * (float)head / 8.0f)) * 1.4426950408889634f;

  const _Float16* __restrict__ Qb = Qh + (size_t)bh * NTOK * 32;
  const _Float16* __restrict__ Kb = Kh + (size_t)bh * NTOK * 32;
  const _Float16* __restrict__ Vb = Vt + (size_t)bh * HEAD_DIM * NTOK;

  // cooperative staging: 16 bytes per thread per chunk (4KB total per chunk)
  const int  sk   = tid & 127;
  const int  srow = sk >> 2;                 // 0..31
  const int  soff = (sk & 3) << 3;           // 0,8,16,24 halves
  const bool doV  = tid >= 128;

  auto stage = [&](int kc, int buf) {
    const _Float16* src = doV ? (Vb + (size_t)srow * NTOK + kc + soff)
                              : (Kb + (size_t)(kc + srow) * 32 + soff);
    _Float16* dst = doV ? &Vc[buf][srow][soff] : &Kc[buf][srow][soff];
#ifdef USE_ASYNC
    __builtin_amdgcn_global_load_async_to_lds_b128(
        (gv4i_p)src, (sv4i_p)dst, 0, 0);
#else
    *(v8h*)dst = *(const v8h*)src;
#endif
  };

  const v16h qa = frag_row(Qb + (size_t)(q0 + ln) * 32, lane);

  v8f acc0 = {}, acc1 = {};
  float lrow[8] = {0, 0, 0, 0, 0, 0, 0, 0};  // lane-local partial row sums
  float fqh[8], fqw[8];
  #pragma unroll
  for (int r = 0; r < 8; ++r) {
    const int qp = q0 + r + 8 * hi;
    fqh[r] = (float)(qp / IMG); fqw[r] = (float)(qp % IMG);
  }
  // incremental key coordinates (pos = kc + ln, kc + 16 + ln)
  int kh0i = 0, kw0i = ln;
  int kh1i = 0, kw1i = 16 + ln;

  stage(0, 0);
  for (int c = 0; c < NCHUNK; ++c) {
    if (c + 1 < NCHUNK) stage((c + 1) * 32, (c + 1) & 1);
#ifdef USE_ASYNC
    if (c + 1 < NCHUNK) __builtin_amdgcn_s_wait_asynccnt(1);
    else                __builtin_amdgcn_s_wait_asynccnt(0);
#endif
    __syncthreads();                         // chunk c visible to all waves
    const int cur = c & 1;

    const v16h kb0 = frag_row(&Kc[cur][ln][0], lane);
    const v16h kb1 = frag_row(&Kc[cur][16 + ln][0], lane);
    v8f s0 = {}, s1 = {};
    s0 = __builtin_amdgcn_wmma_f32_16x16x32_f16(false, qa, false, kb0,
                                                (short)0, s0, false, false);
    s1 = __builtin_amdgcn_wmma_f32_16x16x32_f16(false, qa, false, kb1,
                                                (short)0, s1, false, false);

    const float fkh0 = (float)kh0i, fkw0 = (float)kw0i;
    const float fkh1 = (float)kh1i, fkw1 = (float)kw1i;
    #pragma unroll
    for (int r = 0; r < 8; ++r) {
      const float d0 = fabsf(fqh[r] - fkh0) + fabsf(fqw[r] - fkw0);
      const float d1 = fabsf(fqh[r] - fkh1) + fabsf(fqw[r] - fkw1);
      const float p0 = exp2f(s0[r] + d0 * decay2);   // logits bounded: no max
      const float p1 = exp2f(s1[r] + d1 * decay2);
      lrow[r] += p0 + p1;
      pbuf[wid][r + 8 * hi][ln]      = (_Float16)p0;
      pbuf[wid][r + 8 * hi][16 + ln] = (_Float16)p1;
    }
    kw0i += 32; if (kw0i >= IMG) { kw0i -= IMG; ++kh0i; }
    kw1i += 32; if (kw1i >= IMG) { kw1i -= IMG; ++kh1i; }

    const v16h pa  = frag_row(&pbuf[wid][ln][0], lane);   // 2x ds_load_b128
    const v16h vb0 = frag_row(&Vc[cur][ln][0], lane);
    const v16h vb1 = frag_row(&Vc[cur][16 + ln][0], lane);
    acc0 = __builtin_amdgcn_wmma_f32_16x16x32_f16(false, pa, false, vb0,
                                                  (short)0, acc0, false, false);
    acc1 = __builtin_amdgcn_wmma_f32_16x16x32_f16(false, pa, false, vb1,
                                                  (short)0, acc1, false, false);
    __syncthreads();                         // protect buffers for next stage
  }

  // single deferred row-sum reduction across the 16-lane half
  #pragma unroll
  for (int m = 1; m <= 8; m <<= 1) {
    #pragma unroll
    for (int r = 0; r < 8; ++r)
      lrow[r] += __shfl_xor(lrow[r], m, 32);
  }

  const int headc = head * HEAD_DIM;
  #pragma unroll
  for (int r = 0; r < 8; ++r) {
    const float  inv = 1.0f / lrow[r];
    const int    row = q0 + r + 8 * hi;
    const size_t ob  = ((size_t)bimg * NTOK + row) * 256 + headc;
    Ah[ob + ln]      = (_Float16)(acc0[r] * inv + Lp[ob + ln]);
    Ah[ob + 16 + ln] = (_Float16)(acc1[r] * inv + Lp[ob + 16 + ln]);
  }
}

// ---------------------------------------------------------------------------
// Kernel 4: out = Ah @ Wo + bo   (4608 x 256, K=256), 16x32 tile per wave.
// ---------------------------------------------------------------------------
__global__ __launch_bounds__(256) void out_proj_kernel(
    const _Float16* __restrict__ Ah, const _Float16* __restrict__ WoT,
    const float* __restrict__ bo, float* __restrict__ Out)
{
  const int lane  = threadIdx.x & 31;
  const int wid   = threadIdx.x >> 5;
  const int wtile = blockIdx.x * 8 + wid;   // 0..2303
  const int tn    = wtile & 7;
  const int tm    = wtile >> 3;
  const int m0    = tm * 16;
  const int n0    = tn * 32;
  const int ln    = lane & 15;
  const int hi    = (lane >> 4) & 1;

  const _Float16* __restrict__ arow  = Ah + (size_t)(m0 + ln) * 256;
  const _Float16* __restrict__ bcol0 = WoT + (size_t)(n0 + ln) * 256;
  const _Float16* __restrict__ bcol1 = WoT + (size_t)(n0 + 16 + ln) * 256;

  v8f acc0 = {}, acc1 = {};
  #pragma unroll
  for (int k0 = 0; k0 < 256; k0 += 32) {
    const v16h a  = frag_row(arow + k0, lane);
    const v16h b0 = frag_row(bcol0 + k0, lane);
    const v16h b1 = frag_row(bcol1 + k0, lane);
    acc0 = __builtin_amdgcn_wmma_f32_16x16x32_f16(false, a, false, b0,
                                                  (short)0, acc0, false, false);
    acc1 = __builtin_amdgcn_wmma_f32_16x16x32_f16(false, a, false, b1,
                                                  (short)0, acc1, false, false);
  }
  const float bv0 = bo[n0 + ln];
  const float bv1 = bo[n0 + 16 + ln];
  #pragma unroll
  for (int r = 0; r < 8; ++r) {
    const size_t row = m0 + r + 8 * hi;
    Out[row * 256 + n0 + ln]      = acc0[r] + bv0;
    Out[row * 256 + n0 + 16 + ln] = acc1[r] + bv1;
  }
}

// ---------------------------------------------------------------------------
extern "C" void kernel_launch(void* const* d_in, const int* in_sizes, int n_in,
                              void* d_out, int out_size, void* d_ws, size_t ws_size,
                              hipStream_t stream) {
  const float* X  = (const float*)d_in[0];
  const float* Wq = (const float*)d_in[1];
  const float* bq = (const float*)d_in[2];
  const float* Wk = (const float*)d_in[3];
  const float* bk = (const float*)d_in[4];
  const float* Wv = (const float*)d_in[5];
  const float* bv = (const float*)d_in[6];
  const float* lw = (const float*)d_in[7];
  const float* lb = (const float*)d_in[8];
  const float* Wo = (const float*)d_in[9];
  const float* bo = (const float*)d_in[10];
  float* out = (float*)d_out;

  // workspace carve-up (all 16B-aligned)
  char* p = (char*)d_ws;
  _Float16* Xh  = (_Float16*)p;            p += (size_t)MATSZ * 2;
  _Float16* Wt4 = (_Float16*)p;            p += (size_t)4 * 65536 * 2;
  _Float16* Qh  = (_Float16*)p;            p += (size_t)MATSZ * 2;
  _Float16* Kh  = (_Float16*)p;            p += (size_t)MATSZ * 2;
  _Float16* Vt  = (_Float16*)p;            p += (size_t)MATSZ * 2;
  float*    Vf  = (float*)p;               p += (size_t)MATSZ * 4;
  float*    Lp  = (float*)p;               p += (size_t)MATSZ * 4;
  _Float16* Ah  = (_Float16*)p;            p += (size_t)MATSZ * 2;

  pack_kernel<<<(MATSZ + 4 * 65536) / 256, 256, 0, stream>>>(X, Wq, Wk, Wv, Wo, Xh, Wt4);
  qkv_rotary_kernel<<<864, 256, 0, stream>>>(Xh, Wt4, bq, bk, bv, Qh, Kh, Vf, Vt);
  lepe_kernel<<<MATSZ / 256, 256, 0, stream>>>(Vf, lw, lb, Lp);
  flash_attn_kernel<<<16 * 18, 256, 0, stream>>>(Qh, Kh, Vt, Lp, Ah);
  out_proj_kernel<<<288, 256, 0, stream>>>(Ah, (const _Float16*)(Wt4 + 3 * 65536), bo, out);
}
